// CustomMHA_53369263620184
// MI455X (gfx1250) — matrix operand order repaired and checked by reading, not compiled, not run on hardware
//
#include <hip/hip_runtime.h>
#include <hip/hip_bf16.h>

// ---------------------------------------------------------------------------
// Problem constants (B=2, T=2048, C=1024, H=16, D=64)
// ---------------------------------------------------------------------------
#define B_ 2
#define T_ 2048
#define C_ 1024
#define H_ 16
#define D_ 64
#define BT_ (B_ * T_)          // 4096

typedef __bf16 bf16_t;
typedef bf16_t v16bf __attribute__((ext_vector_type(16)));
typedef bf16_t v8bf  __attribute__((ext_vector_type(8)));
typedef float  v8f   __attribute__((ext_vector_type(8)));

#define WMMA_BF16(a, b, c) \
  __builtin_amdgcn_wmma_f32_16x16x32_bf16(false, (a), false, (b), (short)0, (c), false, false)

// Build a 16-element bf16 fragment from two contiguous 16-byte runs.
static __device__ __forceinline__ v16bf load_frag(const bf16_t* p0, const bf16_t* p1) {
  v16bf r;
  reinterpret_cast<v8bf*>(&r)[0] = *reinterpret_cast<const v8bf*>(p0);
  reinterpret_cast<v8bf*>(&r)[1] = *reinterpret_cast<const v8bf*>(p1);
  return r;
}

// Async global->LDS copy of one 16-byte segment (gfx1250, tracked by ASYNCcnt).
// For an LDS-space generic pointer, addr[31:0] is the LDS byte offset.
static __device__ __forceinline__ void async_copy16(const bf16_t* g, bf16_t* l) {
  unsigned lds_off = (unsigned)(uintptr_t)l;
  asm volatile("global_load_async_to_lds_b128 %0, %1, off"
               :: "v"(lds_off), "v"(g) : "memory");
}
static __device__ __forceinline__ void wait_async0() {
  asm volatile("s_wait_asynccnt 0x0" ::: "memory");
}

// ---------------------------------------------------------------------------
// Kernel 0: f32 -> bf16 conversion (activations + weights)
// ---------------------------------------------------------------------------
__global__ __launch_bounds__(256)
void cvt_bf16_kernel(const float* __restrict__ in, bf16_t* __restrict__ out, int n) {
  int i = (blockIdx.x * blockDim.x + threadIdx.x) * 4;
  if (i + 3 < n) {
    float4 v = *reinterpret_cast<const float4*>(in + i);
    out[i + 0] = (bf16_t)v.x;
    out[i + 1] = (bf16_t)v.y;
    out[i + 2] = (bf16_t)v.z;
    out[i + 3] = (bf16_t)v.w;
  }
}

// ---------------------------------------------------------------------------
// Kernel 1: fused head-gathered QKV projection.
//   grid = (BT/128, H, 3)  block = 128 (4 waves)
//   mat 0: Q[b][j][t][d]   mat 1: K[b][j][t][d]   mat 2: V^T[b][j][d][t]
// ---------------------------------------------------------------------------
__global__ __launch_bounds__(128)
void qkv_kernel(const bf16_t* __restrict__ Xb,  const bf16_t* __restrict__ X1b,
                const bf16_t* __restrict__ X2b, const bf16_t* __restrict__ X3b,
                const bf16_t* __restrict__ Wab, const float* __restrict__ b_attn,
                const int* __restrict__ head_idx, const int* __restrict__ head_idxs,
                const int* __restrict__ perm,
                bf16_t* __restrict__ Qb, bf16_t* __restrict__ Kb, bf16_t* __restrict__ Vtb)
{
  const int mat  = blockIdx.z;          // 0=q 1=k 2=v
  const int j    = blockIdx.y;          // final head slot
  const int wave = threadIdx.x >> 5;
  const int lane = threadIdx.x & 31;
  const int lm   = lane & 15;
  const int lh   = lane >> 4;

  // Head gather: permutation slot 0 -> x1/x2/x3 with head_idx, else x with head_idxs[c-1]
  const int c = perm[j];
  int head;
  const bf16_t* src;
  if (c == 0) {
    head = head_idx[0];
    src  = (mat == 0) ? X1b : (mat == 1) ? X2b : X3b;
  } else {
    head = head_idxs[c - 1];
    src  = Xb;
  }
  const int woff = mat * C_ + head * D_;          // row offset into W_attn (3C x C)
  const bf16_t* wbase = Wab + (size_t)woff * C_;

  const int mrow0 = blockIdx.x * 128 + wave * 32;

  v8f acc[2][4];
  #pragma unroll
  for (int mt = 0; mt < 2; ++mt)
    #pragma unroll
    for (int nt = 0; nt < 4; ++nt)
      acc[mt][nt] = (v8f)0.0f;

  for (int k0 = 0; k0 < C_; k0 += 32) {
    // B fragments: W^T tile, column n = woff+n row of W, contiguous along k.
    v16bf bfr[4];
    #pragma unroll
    for (int nt = 0; nt < 4; ++nt) {
      const bf16_t* wp = wbase + (size_t)(nt * 16 + lm) * C_ + k0 + lh * 16;
      bfr[nt] = load_frag(wp, wp + 8);
    }
    // A fragments: X rows, contiguous along k (k = 16h + s).
    v16bf afr[2];
    #pragma unroll
    for (int mt = 0; mt < 2; ++mt) {
      const bf16_t* ap = src + (size_t)(mrow0 + mt * 16 + lm) * C_ + k0 + lh * 16;
      afr[mt] = load_frag(ap, ap + 8);
    }
    #pragma unroll
    for (int mt = 0; mt < 2; ++mt)
      #pragma unroll
      for (int nt = 0; nt < 4; ++nt)
        acc[mt][nt] = WMMA_BF16(afr[mt], bfr[nt], acc[mt][nt]);
  }

  const int bb   = mrow0 >> 11;       // / T_
  const int trow = mrow0 & (T_ - 1);
  const size_t hoff = (size_t)bb * H_ + j;

  // Branch-free epilogue: pick destination base + strides once.
  bf16_t* dst;
  int rs, ns;   // element strides: per local row (t), per column (n)
  if (mat == 0)      { dst = Qb  + (hoff * T_ + trow) * D_;      rs = D_; ns = 1;  }
  else if (mat == 1) { dst = Kb  + (hoff * T_ + trow) * D_;      rs = D_; ns = 1;  }
  else               { dst = Vtb +  hoff * D_ * T_ + trow;       rs = 1;  ns = T_; }

  #pragma unroll
  for (int nt = 0; nt < 4; ++nt) {
    const int n = nt * 16 + lm;
    const float bias = b_attn[woff + n];
    #pragma unroll
    for (int mt = 0; mt < 2; ++mt) {
      #pragma unroll
      for (int r = 0; r < 8; ++r) {
        const int tl = mt * 16 + lh * 8 + r;
        dst[(size_t)tl * rs + (size_t)n * ns] = (bf16_t)(acc[mt][nt][r] + bias);
      }
    }
  }
}

// ---------------------------------------------------------------------------
// Kernel 2: flash attention (online softmax), per (b, head).
//   grid = (T/64, H, B)  block = 128 (4 waves, each owns 16 query rows)
//   K/V chunks (32 keys) staged into LDS with async copies, double-buffered,
//   shared by all 4 waves (4x traffic reduction + copy/compute overlap).
// ---------------------------------------------------------------------------
__global__ __launch_bounds__(128)
void attn_kernel(const bf16_t* __restrict__ Qb, const bf16_t* __restrict__ Kb,
                 const bf16_t* __restrict__ Vtb, bf16_t* __restrict__ Yb)
{
  __shared__ bf16_t Kls[2][32 * 64];    // [buf][kk][d]   4KB each
  __shared__ bf16_t Vls[2][64 * 32];    // [buf][d][kk]   4KB each
  __shared__ bf16_t Plds[4][16 * 32];   // per-wave P staging

  const int bz   = blockIdx.z;
  const int j    = blockIdx.y;
  const int tid  = threadIdx.x;
  const int wave = tid >> 5;
  const int lane = tid & 31;
  const int lm   = lane & 15;
  const int lh   = lane >> 4;
  const int qbase = blockIdx.x * 64 + wave * 16;

  const size_t hoff = (size_t)bz * H_ + j;
  const bf16_t* Qh = Qb  + hoff * T_ * D_;
  const bf16_t* Kh = Kb  + hoff * T_ * D_;
  const bf16_t* Vh = Vtb + hoff * D_ * T_;

  // Issue async staging of one 32-key chunk into LDS buffer `buf`.
  auto stage = [&](int kk0, int buf) {
    // K chunk: 32x64 bf16 = 4KB, fully contiguous in global.
    const bf16_t* gk = Kh + (size_t)kk0 * D_;
    bf16_t* lk = &Kls[buf][0];
    async_copy16(gk + tid * 8,         lk + tid * 8);
    async_copy16(gk + (tid + 128) * 8, lk + (tid + 128) * 8);
    // V^T chunk: 64 rows of 32 contiguous elems (4 x 16B segments per row).
    bf16_t* lv = &Vls[buf][0];
    const int s0 = tid, s1 = tid + 128;
    async_copy16(Vh + (size_t)(s0 >> 2) * T_ + kk0 + (s0 & 3) * 8, lv + s0 * 8);
    async_copy16(Vh + (size_t)(s1 >> 2) * T_ + kk0 + (s1 & 3) * 8, lv + s1 * 8);
  };

  // Q as two A-fragments (d 0..31, 32..63), resident for the whole K loop.
  v16bf qa[2];
  {
    const bf16_t* qp = Qh + (size_t)(qbase + lm) * D_ + lh * 16;
    qa[0] = load_frag(qp,      qp + 8);
    qa[1] = load_frag(qp + 32, qp + 40);
  }

  float m_i[8], l_i[8];
  v8f o[4];
  #pragma unroll
  for (int r = 0; r < 8; ++r) { m_i[r] = -1e30f; l_i[r] = 0.0f; }
  #pragma unroll
  for (int nt = 0; nt < 4; ++nt) o[nt] = (v8f)0.0f;

  const float scale = 0.125f;           // 1/sqrt(64)

  // Prologue: stage chunk 0.
  stage(0, 0);
  wait_async0();
  __syncthreads();

  int buf = 0;
  for (int kk0 = 0; kk0 < T_; kk0 += 32) {
    // Kick off the next chunk while computing on this one.
    if (kk0 + 32 < T_) stage(kk0 + 32, buf ^ 1);

    const bf16_t* kbuf = &Kls[buf][0];
    const bf16_t* vbuf = &Vls[buf][0];

    // S = Q * K^T for a 16q x 32k chunk  (4 wmma), K fragments from LDS.
    v8f s[2];
    s[0] = (v8f)0.0f; s[1] = (v8f)0.0f;
    #pragma unroll
    for (int kt = 0; kt < 2; ++kt) {
      const bf16_t* kp = kbuf + (kt * 16 + lm) * D_ + lh * 16;
      v16bf kb0 = load_frag(kp,      kp + 8);    // d 0..31
      v16bf kb1 = load_frag(kp + 32, kp + 40);   // d 32..63
      s[kt] = WMMA_BF16(qa[0], kb0, s[kt]);
      s[kt] = WMMA_BF16(qa[1], kb1, s[kt]);
    }

    // Online softmax statistics (rows across VGPR index, cols across 16 lanes).
    float cmax[8];
    #pragma unroll
    for (int r = 0; r < 8; ++r) {
      s[0][r] *= scale; s[1][r] *= scale;
      cmax[r] = fmaxf(s[0][r], s[1][r]);
    }
    #pragma unroll
    for (int off = 1; off < 16; off <<= 1)
      #pragma unroll
      for (int r = 0; r < 8; ++r)
        cmax[r] = fmaxf(cmax[r], __shfl_xor(cmax[r], off, 32));

    float alpha[8], csum[8];
    #pragma unroll
    for (int r = 0; r < 8; ++r) {
      const float mn = fmaxf(m_i[r], cmax[r]);
      alpha[r] = __expf(m_i[r] - mn);
      s[0][r]  = __expf(s[0][r] - mn);
      s[1][r]  = __expf(s[1][r] - mn);
      csum[r]  = s[0][r] + s[1][r];
      m_i[r]   = mn;
    }
    #pragma unroll
    for (int off = 1; off < 16; off <<= 1)
      #pragma unroll
      for (int r = 0; r < 8; ++r)
        csum[r] += __shfl_xor(csum[r], off, 32);
    #pragma unroll
    for (int r = 0; r < 8; ++r)
      l_i[r] = l_i[r] * alpha[r] + csum[r];
    #pragma unroll
    for (int nt = 0; nt < 4; ++nt)
      #pragma unroll
      for (int r = 0; r < 8; ++r)
        o[nt][r] *= alpha[r];

    // P: C-layout f32 -> LDS -> A-layout bf16 (per-wave region, in-order DS).
    bf16_t* pl = &Plds[wave][0];
    #pragma unroll
    for (int kt = 0; kt < 2; ++kt)
      #pragma unroll
      for (int r = 0; r < 8; ++r)
        pl[(lh * 8 + r) * 32 + kt * 16 + lm] = (bf16_t)s[kt][r];
    asm volatile("s_wait_dscnt 0x0" ::: "memory");
    const bf16_t* pr = pl + lm * 32 + lh * 16;
    v16bf pa = load_frag(pr, pr + 8);

    // O += P * V  (V^T in LDS -> contiguous B fragments)
    #pragma unroll
    for (int nt = 0; nt < 4; ++nt) {
      const bf16_t* vp = vbuf + (nt * 16 + lm) * 32 + lh * 16;
      v16bf vb = load_frag(vp, vp + 8);
      o[nt] = WMMA_BF16(pa, vb, o[nt]);
    }

    // Next buffer must be fully staged (own wave) and everyone done reading.
    wait_async0();
    __syncthreads();
    buf ^= 1;
  }

  // Normalize and scatter back to [b][t][j*64+d] bf16 for the projection GEMM.
  float inv[8];
  #pragma unroll
  for (int r = 0; r < 8; ++r) inv[r] = 1.0f / l_i[r];
  #pragma unroll
  for (int nt = 0; nt < 4; ++nt) {
    #pragma unroll
    for (int r = 0; r < 8; ++r) {
      const int t   = qbase + lh * 8 + r;
      const int col = j * D_ + nt * 16 + lm;
      Yb[((size_t)bz * T_ + t) * C_ + col] = (bf16_t)(o[nt][r] * inv[r]);
    }
  }
}

// ---------------------------------------------------------------------------
// Kernel 3: output projection  out = Y @ W_proj^T + b_proj  (f32 out)
//   grid = (BT/128, C/64)  block = 128 (4 waves)
// ---------------------------------------------------------------------------
__global__ __launch_bounds__(128)
void proj_kernel(const bf16_t* __restrict__ Yb, const bf16_t* __restrict__ Wpb,
                 const float* __restrict__ b_proj, float* __restrict__ out)
{
  const int wave = threadIdx.x >> 5;
  const int lane = threadIdx.x & 31;
  const int lm   = lane & 15;
  const int lh   = lane >> 4;
  const int mrow0 = blockIdx.x * 128 + wave * 32;
  const int ncol0 = blockIdx.y * 64;

  v8f acc[2][4];
  #pragma unroll
  for (int mt = 0; mt < 2; ++mt)
    #pragma unroll
    for (int nt = 0; nt < 4; ++nt)
      acc[mt][nt] = (v8f)0.0f;

  for (int k0 = 0; k0 < C_; k0 += 32) {
    v16bf bfr[4];
    #pragma unroll
    for (int nt = 0; nt < 4; ++nt) {
      const bf16_t* wp = Wpb + (size_t)(ncol0 + nt * 16 + lm) * C_ + k0 + lh * 16;
      bfr[nt] = load_frag(wp, wp + 8);
    }
    v16bf afr[2];
    #pragma unroll
    for (int mt = 0; mt < 2; ++mt) {
      const bf16_t* ap = Yb + (size_t)(mrow0 + mt * 16 + lm) * C_ + k0 + lh * 16;
      afr[mt] = load_frag(ap, ap + 8);
    }
    #pragma unroll
    for (int mt = 0; mt < 2; ++mt)
      #pragma unroll
      for (int nt = 0; nt < 4; ++nt)
        acc[mt][nt] = WMMA_BF16(afr[mt], bfr[nt], acc[mt][nt]);
  }

  #pragma unroll
  for (int nt = 0; nt < 4; ++nt) {
    const int n = ncol0 + nt * 16 + lm;
    const float bias = b_proj[n];
    #pragma unroll
    for (int mt = 0; mt < 2; ++mt)
      #pragma unroll
      for (int r = 0; r < 8; ++r)
        out[(size_t)(mrow0 + mt * 16 + lh * 8 + r) * C_ + n] = acc[mt][nt][r] + bias;
  }
}

// ---------------------------------------------------------------------------
// Launch
// ---------------------------------------------------------------------------
extern "C" void kernel_launch(void* const* d_in, const int* in_sizes, int n_in,
                              void* d_out, int out_size, void* d_ws, size_t ws_size,
                              hipStream_t stream) {
  const float* x1     = (const float*)d_in[0];
  const float* x2     = (const float*)d_in[1];
  const float* x3     = (const float*)d_in[2];
  const float* x      = (const float*)d_in[3];
  const float* W_attn = (const float*)d_in[4];
  const float* b_attn = (const float*)d_in[5];
  const float* W_proj = (const float*)d_in[6];
  const float* b_proj = (const float*)d_in[7];
  const int* head_idx  = (const int*)d_in[8];
  const int* head_idxs = (const int*)d_in[9];
  const int* perm      = (const int*)d_in[10];
  float* out = (float*)d_out;

  // Workspace layout (bytes): 4x act bf16 (8MB ea) | Wab 6MB | Wpb 2MB |
  // Q 8MB | K 8MB | V^T 8MB | Y 8MB  = 72 MB total
  char* ws = (char*)d_ws;
  const size_t MB = 1u << 20;
  bf16_t* Xb  = (bf16_t*)(ws +  0 * MB);
  bf16_t* X1b = (bf16_t*)(ws +  8 * MB);
  bf16_t* X2b = (bf16_t*)(ws + 16 * MB);
  bf16_t* X3b = (bf16_t*)(ws + 24 * MB);
  bf16_t* Wab = (bf16_t*)(ws + 32 * MB);
  bf16_t* Wpb = (bf16_t*)(ws + 38 * MB);
  bf16_t* Qb  = (bf16_t*)(ws + 40 * MB);
  bf16_t* Kb  = (bf16_t*)(ws + 48 * MB);
  bf16_t* Vtb = (bf16_t*)(ws + 56 * MB);
  bf16_t* Yb  = (bf16_t*)(ws + 64 * MB);

  const int nAct = BT_ * C_;        // 4194304
  const int nWa  = 3 * C_ * C_;     // 3145728
  const int nWp  = C_ * C_;         // 1048576
  auto cvtGrid = [](int n) { return dim3((unsigned)((n / 4 + 255) / 256)); };

  cvt_bf16_kernel<<<cvtGrid(nAct), 256, 0, stream>>>(x,      Xb,  nAct);
  cvt_bf16_kernel<<<cvtGrid(nAct), 256, 0, stream>>>(x1,     X1b, nAct);
  cvt_bf16_kernel<<<cvtGrid(nAct), 256, 0, stream>>>(x2,     X2b, nAct);
  cvt_bf16_kernel<<<cvtGrid(nAct), 256, 0, stream>>>(x3,     X3b, nAct);
  cvt_bf16_kernel<<<cvtGrid(nWa),  256, 0, stream>>>(W_attn, Wab, nWa);
  cvt_bf16_kernel<<<cvtGrid(nWp),  256, 0, stream>>>(W_proj, Wpb, nWp);

  qkv_kernel<<<dim3(BT_ / 128, H_, 3), 128, 0, stream>>>(
      Xb, X1b, X2b, X3b, Wab, b_attn, head_idx, head_idxs, perm, Qb, Kb, Vtb);

  attn_kernel<<<dim3(T_ / 64, H_, B_), 128, 0, stream>>>(Qb, Kb, Vtb, Yb);

  proj_kernel<<<dim3(BT_ / 128, C_ / 64), 128, 0, stream>>>(Yb, Wpb, b_proj, out);
}
